// TELECHATAttention_11484742549757
// MI455X (gfx1250) — matrix operand order repaired and checked by reading, not compiled, not run on hardware
//
#include <hip/hip_runtime.h>

// ---------------------------------------------------------------------------
// TeleChat attention for MI455X (gfx1250, wave32, WMMA).
// - all matmuls: v_wmma_f32_16x16x32_bf16 (fp32 accumulate)
// - flash-attention tiles staged with BOTH CDNA5 DMA paths, double-buffered:
//     K tile: TDM tensor_load_to_lds (D# descriptor, LDS pad feature,
//             TENSORcnt), one issue per workgroup per tile
//     V tile: GLOBAL_LOAD_ASYNC_TO_LDS_B128 (ASYNCcnt), per-lane
// - GEMMs: 128x128x32 tiles, register-prefetch + packed ds_store_b128,
//   double-buffered LDS so global loads overlap WMMA
// ---------------------------------------------------------------------------

typedef __attribute__((ext_vector_type(16))) __bf16   v16bf;
typedef __attribute__((ext_vector_type(8)))  float    v8f;
typedef __attribute__((ext_vector_type(4)))  unsigned v4u;
typedef __attribute__((ext_vector_type(8)))  int      v8i;

union BfVec { v16bf v; unsigned short u[16]; };

__device__ __forceinline__ float bf2f(unsigned short u) {
  return __uint_as_float(((unsigned)u) << 16);
}
__device__ __forceinline__ unsigned short f2bf(float f) {
  unsigned u = __float_as_uint(f);
  unsigned r = u + 0x7FFFu + ((u >> 16) & 1u);   // round-to-nearest-even
  return (unsigned short)(r >> 16);
}
__device__ __forceinline__ unsigned pack2(float lo, float hi) {
  return (unsigned)f2bf(lo) | ((unsigned)f2bf(hi) << 16);
}

__device__ __forceinline__ v8f wmma_bf16(const BfVec& a, const BfVec& b, v8f c) {
  return __builtin_amdgcn_wmma_f32_16x16x32_bf16(false, a.v, false, b.v,
                                                 (short)0, c, false, false);
}

// CDNA5 async copy: each lane moves 16B global->LDS, tracked by ASYNCcnt.
__device__ __forceinline__ void async_b128(void* lds, const void* gsrc) {
  asm volatile("global_load_async_to_lds_b128 %0, %1, off"
               :: "v"((unsigned)(uintptr_t)lds), "v"(gsrc)
               : "memory");
}
__device__ __forceinline__ void wait_async0() {
  asm volatile("s_wait_asynccnt 0" ::: "memory");
}

// CDNA5 Tensor Data Mover: D# groups 0 (4 SGPRs) + 1 (8 SGPRs), 2D tile.
__device__ __forceinline__ void tdm_load(v4u g0, v8i g1) {
  asm volatile("tensor_load_to_lds %0, %1" :: "s"(g0), "s"(g1) : "memory");
}
__device__ __forceinline__ void wait_tensor0() {
  __builtin_amdgcn_s_wait_tensorcnt(0);
}

static constexpr int S  = 2048;
static constexpr int D  = 4096;
static constexpr int H  = 32;
static constexpr int Dh = 128;

// ---------------------------------------------------------------------------
// Tiled GEMM: C[M,N] = A[M,K] * W[K,N] + bias.  Block tile 128x128, BK=32,
// 256 threads = 8 wave32s (4 row-groups x 2 col-groups); each wave owns
// 32x64 -> 2x4 fp32 accumulators, 8 WMMAs per K-step.  Software pipelined:
// global->regs prefetch for step k+1 in flight during the WMMAs of step k;
// fp32->bf16 convert + packed b128 LDS stores; double-buffered LDS.
// MODE 0: A fp32, epilogue scatters bf16 into head-split Q/K/V [H][S][Dh].
// MODE 1: A bf16, epilogue writes fp32 linear C.
// ---------------------------------------------------------------------------
template <int MODE>
__global__ __launch_bounds__(256)
void gemm_bf16_wmma(const void* __restrict__ Aopq, const float* __restrict__ W,
                    const float* __restrict__ bias, float* __restrict__ Cf,
                    unsigned short* __restrict__ Qs, unsigned short* __restrict__ Ks,
                    unsigned short* __restrict__ Vs, int M, int N, int K) {
  constexpr int BM = 128, BN = 128, BK = 32;
  __shared__ unsigned short As[2][BM][BK + 8];
  __shared__ unsigned short Bs[2][BK][BN + 8];

  const int tid  = threadIdx.x;
  const int lane = tid & 31;
  const int w    = tid >> 5;
  const int wm   = w & 3;
  const int wn   = w >> 2;
  const int bm   = blockIdx.y * BM;
  const int bn   = blockIdx.x * BN;

  v8f acc[2][4] = {};

  const int arow = tid >> 1, acol = (tid & 1) * 16;
  const int brow = tid >> 3, bcol = (tid & 7) * 16;

  const float*    Af = (const float*)Aopq;
  const unsigned* Au = (const unsigned*)Aopq;   // bf16 pairs (MODE 1)

  float    ar[16];
  unsigned au[8];
  float    br[16];

  auto loadregs = [&](int k0) {
    if (MODE == 0) {
      const float* s = Af + (size_t)(bm + arow) * K + k0 + acol;
#pragma unroll
      for (int j = 0; j < 16; ++j) ar[j] = s[j];
    } else {
      const unsigned* s = Au + ((size_t)(bm + arow) * K + k0 + acol) / 2;
#pragma unroll
      for (int j = 0; j < 8; ++j) au[j] = s[j];
    }
    const float* s = W + (size_t)(k0 + brow) * N + bn + bcol;
#pragma unroll
    for (int j = 0; j < 16; ++j) br[j] = s[j];
  };

  auto storelds = [&](int buf) {
    uint4 t0, t1;
    if (MODE == 0) {
      t0.x = pack2(ar[0], ar[1]);   t0.y = pack2(ar[2], ar[3]);
      t0.z = pack2(ar[4], ar[5]);   t0.w = pack2(ar[6], ar[7]);
      t1.x = pack2(ar[8], ar[9]);   t1.y = pack2(ar[10], ar[11]);
      t1.z = pack2(ar[12], ar[13]); t1.w = pack2(ar[14], ar[15]);
    } else {
      t0.x = au[0]; t0.y = au[1]; t0.z = au[2]; t0.w = au[3];
      t1.x = au[4]; t1.y = au[5]; t1.z = au[6]; t1.w = au[7];
    }
    *(uint4*)&As[buf][arow][acol]     = t0;
    *(uint4*)&As[buf][arow][acol + 8] = t1;
    uint4 u0, u1;
    u0.x = pack2(br[0], br[1]);   u0.y = pack2(br[2], br[3]);
    u0.z = pack2(br[4], br[5]);   u0.w = pack2(br[6], br[7]);
    u1.x = pack2(br[8], br[9]);   u1.y = pack2(br[10], br[11]);
    u1.z = pack2(br[12], br[13]); u1.w = pack2(br[14], br[15]);
    *(uint4*)&Bs[buf][brow][bcol]     = u0;
    *(uint4*)&Bs[buf][brow][bcol + 8] = u1;
  };

  loadregs(0);
  const int khi = (lane & 16) >> 1;
  const int kb  = lane & 16;

  int it = 0;
  for (int k0 = 0; k0 < K; k0 += BK, ++it) {
    const int buf = it & 1;
    storelds(buf);
    __syncthreads();
    if (k0 + BK < K) loadregs(k0 + BK);   // in flight during WMMAs below

    BfVec a0, a1;
    const int am = wm * 32 + (lane & 15);
#pragma unroll
    for (int e = 0; e < 16; ++e) {
      const int kk = e + kb;   // contiguous K order: k = 16*(lane/16) + e
      a0.u[e] = As[buf][am][kk];
      a1.u[e] = As[buf][am + 16][kk];
    }
#pragma unroll
    for (int ct = 0; ct < 4; ++ct) {
      BfVec b;
      const int bc = wn * 64 + ct * 16 + (lane & 15);
#pragma unroll
      for (int e = 0; e < 16; ++e) b.u[e] = Bs[buf][kb + e][bc];
      acc[0][ct] = wmma_bf16(a0, b, acc[0][ct]);
      acc[1][ct] = wmma_bf16(a1, b, acc[1][ct]);
    }
  }

  const int r0 = (lane & 16) >> 1;
#pragma unroll
  for (int sub = 0; sub < 2; ++sub) {
#pragma unroll
    for (int ct = 0; ct < 4; ++ct) {
      const int col = bn + wn * 64 + ct * 16 + (lane & 15);
      const float bv = bias[col];
#pragma unroll
      for (int rr = 0; rr < 8; ++rr) {
        const int row = bm + wm * 32 + sub * 16 + rr + r0;
        const float v = acc[sub][ct][rr] + bv;
        if (MODE == 1) {
          Cf[(size_t)row * N + col] = v;
        } else {
          const int part   = col >> 12;        // 0:q 1:k 2:v
          const int within = col & (D - 1);
          const int hh = within >> 7, dd = within & (Dh - 1);
          unsigned short* dst = (part == 0) ? Qs : (part == 1) ? Ks : Vs;
          dst[((size_t)hh * S + row) * Dh + dd] = f2bf(v);
        }
      }
    }
  }
}

// ---------------------------------------------------------------------------
// In-place RoPE on head-split Q/K (theta=10000, freqs concatenated twice).
// ---------------------------------------------------------------------------
__global__ __launch_bounds__(64)
void rope_inplace(unsigned short* __restrict__ Qb, unsigned short* __restrict__ Kb) {
  const int s = blockIdx.x, h = blockIdx.y, i = threadIdx.x;
  const float invf = __powf(10000.0f, -(float)(2 * i) / (float)Dh);
  float sn, cs;
  __sincosf((float)s * invf, &sn, &cs);
  const size_t ob = ((size_t)h * S + s) * Dh;

  const float q1 = bf2f(Qb[ob + i]), q2 = bf2f(Qb[ob + i + 64]);
  Qb[ob + i]      = f2bf(q1 * cs - q2 * sn);
  Qb[ob + i + 64] = f2bf(q2 * cs + q1 * sn);

  const float k1 = bf2f(Kb[ob + i]), k2 = bf2f(Kb[ob + i + 64]);
  Kb[ob + i]      = f2bf(k1 * cs - k2 * sn);
  Kb[ob + i + 64] = f2bf(k2 * cs + k1 * sn);
}

// ---------------------------------------------------------------------------
// Causal flash attention. 256 threads = 8 wave32s, one head, 128 query rows
// (16 per wave). Key tiles of 32 double-buffered:
//   K tile: one TDM tensor_load_to_lds issued by wave 0. The D# pad feature
//           (pad every 64 DWORDs by 4 DWORDs) reproduces the 136-ushort
//           padded LDS row stride so K-fragment ds_load_b128 reads spread
//           across banks.
//   V tile: per-lane global_load_async_to_lds_b128.
// Copy of tile kt+1 overlaps the 16 WMMAs of tile kt.
// ---------------------------------------------------------------------------
__global__ __launch_bounds__(256)
void flash_attn_wmma(const unsigned short* __restrict__ Qb,
                     const unsigned short* __restrict__ Kb,
                     const unsigned short* __restrict__ Vb,
                     unsigned short* __restrict__ Att) {
  __shared__ unsigned short Kt[2][32][Dh + 8];
  __shared__ unsigned short Vt[2][32][Dh + 8];
  __shared__ unsigned short Pt[8][16][32];

  const int tid = threadIdx.x, lane = tid & 31, w = tid >> 5;
  const int q0 = blockIdx.x * 128;
  const int h  = blockIdx.y;
  const unsigned short* Kh = Kb + (size_t)h * S * Dh;
  const unsigned short* Vh = Vb + (size_t)h * S * Dh;

  // D# group 1 (constant): workgroup_mask=0, data_size=2B, pad_enable,
  // pad_interval=5 (64 DWORDs = one 256B row), pad_amount=3 (4 DWORDs),
  // tensor_dim0=128, tensor_dim1=32, tile_dim0=128, tile_dim1=32,
  // tensor_dim0_stride=128.
  v8i tg1;
  tg1[0] = (1 << 16) | (1 << 20) | (5 << 22) | (3 << 25);
  tg1[1] = 128 << 16;
  tg1[2] = 32 << 16;
  tg1[3] = 128 << 16;
  tg1[4] = 32;
  tg1[5] = 128;
  tg1[6] = 0;
  tg1[7] = 0;

  auto issue_k_tdm = [&](int buf, int kt) {    // call from ONE wave only
    const unsigned long long ga =
        (unsigned long long)(uintptr_t)(Kh + (size_t)kt * 32 * Dh);
    v4u g0;
    g0[0] = 1u;                                 // count=1 (valid user D#)
    g0[1] = (unsigned)(uintptr_t)&Kt[buf][0][0];// lds_addr
    g0[2] = (unsigned)ga;                       // global_addr[31:0]
    g0[3] = (unsigned)(ga >> 32) | (2u << 30);  // addr[56:32] | type=2
    tdm_load(g0, tg1);
  };

  // V tile loader: 32B per thread = 2 async b128 per buffer.
  const int trow = tid >> 3;
  const int tcol = (tid & 7) * 16;
  auto issue_v_async = [&](int buf, int kt) {
    const unsigned short* vs = Vh + (size_t)(kt * 32 + trow) * Dh + tcol;
    async_b128(&Vt[buf][trow][tcol],     vs);
    async_b128(&Vt[buf][trow][tcol + 8], vs + 8);
  };

  // Q fragments (16 rows x 128) loaded once.
  BfVec qf[4];
  {
    const int qm  = lane & 15;
    const int khi = (lane & 16) >> 1;
    const unsigned short* qrow = Qb + ((size_t)h * S + q0 + w * 16 + qm) * Dh;
#pragma unroll
    for (int dc = 0; dc < 4; ++dc)
#pragma unroll
      for (int e = 0; e < 16; ++e)
        qf[dc].u[e] = qrow[dc * 32 + e + (khi << 1)];
  }

  v8f O[8] = {};
  float mrow[8], lrow[8];
#pragma unroll
  for (int rr = 0; rr < 8; ++rr) { mrow[rr] = -1e30f; lrow[rr] = 0.f; }

  const int hi8   = (lane & 16) >> 1;
  const int kb16  = lane & 16;
  const int numKT = q0 / 32 + 4;              // causal
  const float scale = 0.08838834764831845f;   // 1/sqrt(128)

  if (w == 0) issue_k_tdm(0, 0);              // TDM ignores EXEC; wave-level
  issue_v_async(0, 0);
  wait_async0();
  wait_tensor0();
  __syncthreads();

  for (int kt = 0; kt < numKT; ++kt) {
    const int cur = kt & 1;
    if (kt + 1 < numKT) {                     // overlap copies with compute
      if (w == 0) issue_k_tdm(cur ^ 1, kt + 1);
      issue_v_async(cur ^ 1, kt + 1);
    }

    // Scores: 16x32 = two 16x16 accumulators, contraction over Dh in 4 steps.
    v8f Sa = {}, Sb = {};
#pragma unroll
    for (int dc = 0; dc < 4; ++dc) {
      BfVec b0, b1;
      const int n = lane & 15;
#pragma unroll
      for (int e = 0; e < 16; ++e) {
        b0.u[e] = Kt[cur][n]     [dc * 32 + kb16 + e];
        b1.u[e] = Kt[cur][16 + n][dc * 32 + kb16 + e];
      }
      Sa = wmma_bf16(qf[dc], b0, Sa);
      Sb = wmma_bf16(qf[dc], b1, Sb);
    }

    // Causal mask + online softmax (fp32).
    float pa[8], pb[8];
    const int rowbase = q0 + w * 16 + hi8;
    const int colA = kt * 32 + (lane & 15);
    const int colB = colA + 16;
#pragma unroll
    for (int rr = 0; rr < 8; ++rr) {
      const int row = rowbase + rr;
      float sa = Sa[rr] * scale, sb = Sb[rr] * scale;
      if (colA > row) sa = -1e30f;
      if (colB > row) sb = -1e30f;
      float mx = fmaxf(sa, sb);
#pragma unroll
      for (int off = 8; off; off >>= 1) mx = fmaxf(mx, __shfl_xor(mx, off, 32));
      const float mn = fmaxf(mrow[rr], mx);
      const float alpha = __expf(mrow[rr] - mn);
      const float ea = __expf(sa - mn);
      const float eb = __expf(sb - mn);
      float sum = ea + eb;
#pragma unroll
      for (int off = 8; off; off >>= 1) sum += __shfl_xor(sum, off, 32);
      mrow[rr] = mn;
      lrow[rr] = lrow[rr] * alpha + sum;
      pa[rr] = ea; pb[rr] = eb;
#pragma unroll
      for (int dt = 0; dt < 8; ++dt) O[dt][rr] *= alpha;
    }

    // P (C-layout) -> A-fragment layout via per-wave LDS scratch.
#pragma unroll
    for (int rr = 0; rr < 8; ++rr) {
      Pt[w][rr + hi8][lane & 15]        = f2bf(pa[rr]);
      Pt[w][rr + hi8][16 + (lane & 15)] = f2bf(pb[rr]);
    }
    __syncthreads();

    // PV: O[16,128] += P[16,32] * V[32,128]  (8 WMMAs).
    BfVec pf;
    const int pm = lane & 15;
#pragma unroll
    for (int e = 0; e < 16; ++e) pf.u[e] = Pt[w][pm][e + kb16];
#pragma unroll
    for (int dt = 0; dt < 8; ++dt) {
      BfVec bv;
      const int d = dt * 16 + (lane & 15);
#pragma unroll
      for (int e = 0; e < 16; ++e) bv.u[e] = Vt[cur][kb16 + e][d];
      O[dt] = wmma_bf16(pf, bv, O[dt]);
    }

    wait_async0();       // V tile for kt+1 landed
    wait_tensor0();      // K tile for kt+1 landed (wave 0; 0 elsewhere)
    __syncthreads();
  }

  // Normalize and store merged-head bf16 output [S][D].
#pragma unroll
  for (int rr = 0; rr < 8; ++rr) {
    const float inv = 1.0f / lrow[rr];
    const int row = q0 + w * 16 + hi8 + rr;
#pragma unroll
    for (int dt = 0; dt < 8; ++dt) {
      const int d = dt * 16 + (lane & 15);
      Att[(size_t)row * D + h * Dh + d] = f2bf(O[dt][rr] * inv);
    }
  }
}

// ---------------------------------------------------------------------------
extern "C" void kernel_launch(void* const* d_in, const int* in_sizes, int n_in,
                              void* d_out, int out_size, void* d_ws, size_t ws_size,
                              hipStream_t stream) {
  const float* X  = (const float*)d_in[0];   // hidden_states [1,2048,4096]
  const float* Wa = (const float*)d_in[1];   // w_attn [4096,12288]
  const float* Ba = (const float*)d_in[2];   // b_attn [12288]
  const float* Wp = (const float*)d_in[3];   // w_proj [4096,4096]
  const float* Bp = (const float*)d_in[4];   // b_proj [4096]
  float* out = (float*)d_out;

  // Workspace: Qb|Kb|Vb|Att bf16, 16 MB each -> 64 MB.
  unsigned short* Qb  = (unsigned short*)d_ws;
  unsigned short* Kb  = Qb + (size_t)S * D;
  unsigned short* Vb  = Kb + (size_t)S * D;
  unsigned short* Att = Vb + (size_t)S * D;

  // 1) QKV GEMM, fused bf16 convert + head-split scatter.
  gemm_bf16_wmma<0><<<dim3(3 * D / 128, S / 128), 256, 0, stream>>>(
      X, Wa, Ba, nullptr, Qb, Kb, Vb, S, 3 * D, D);

  // 2) RoPE in place on Q, K.
  rope_inplace<<<dim3(S, H), 64, 0, stream>>>(Qb, Kb);

  // 3) Causal flash attention (TDM K tiles + async V tiles).
  flash_attn_wmma<<<dim3(S / 128, H), 256, 0, stream>>>(Qb, Kb, Vb, Att);

  // 4) Output projection, fp32 out + bias.
  gemm_bf16_wmma<1><<<dim3(D / 128, S / 128), 256, 0, stream>>>(
      Att, Wp, Bp, out, nullptr, nullptr, nullptr, S, D, D);
}
